// Attention_43482248905226
// MI455X (gfx1250) — matrix operand order repaired and checked by reading, not compiled, not run on hardware
//
#include <hip/hip_runtime.h>
#include <hip/hip_bf16.h>

// ---------------------------------------------------------------------------
// MI455X (gfx1250) GQA attention.
//   * all matrix math on v_wmma_f32_16x16x32_bf16 (f32 accum)
//   * all tile staging via the Tensor Data Mover (tensor_load_to_lds),
//     double-buffered, TENSORcnt-synchronized, with TDM LDS padding for
//     conflict-free fragment strides
//   * fp32->bf16 conversion hoisted into one-shot elementwise kernels
// Roofline: ~77 GF vs ~100 MB unique bytes (fits in 192 MB L2) -> matrix-core
// bound, so the GEMM/attention inner loops must be pure ds_load + wmma.
// ---------------------------------------------------------------------------

typedef __attribute__((ext_vector_type(16))) __bf16 v16bf;
typedef __attribute__((ext_vector_type(8)))  __bf16 v8bf;
typedef __attribute__((ext_vector_type(8)))  float  v8f;
typedef __attribute__((ext_vector_type(4)))  unsigned int v4u;
typedef __attribute__((ext_vector_type(8)))  int v8i;
typedef __attribute__((ext_vector_type(4)))  int v4i;

#define DIM      2048
#define LCTX     2048
#define NH       32
#define NKV      8
#define HD       64
#define SCALE    0.125f

// ---------------------------------------------------------------------------
// TDM 2D tile load: global (row-major, stride_elems) -> LDS with padding.
// 2-byte elements. pad_iv: pad after 2<<pad_iv DWORDs; pad_amt: pad_amt+1
// DWORDs inserted. Issued per-wave (EXEC ignored), tracked by TENSORcnt.
// ---------------------------------------------------------------------------
__device__ __forceinline__ unsigned lds_off(const void* p) {
  return (unsigned)(unsigned long long)p;   // LDS aperture: addr[31:0] = offset
}

__device__ __forceinline__ void tdm_load_2d(unsigned lds, const void* gsrc,
                                            unsigned tile_w, unsigned tile_h,
                                            unsigned stride_elems,
                                            unsigned pad_iv, unsigned pad_amt) {
  const unsigned long long ga = (unsigned long long)gsrc;
  const unsigned td0 = stride_elems;     // tensor dim0 (in-bounds by design)
  const unsigned td1 = 1u << 24;         // tensor dim1: huge, no OOB clipping
  const v4u g0 = { 1u,                                   // count=1 user D#
                   lds,
                   (unsigned)ga,
                   (unsigned)((ga >> 32) & 0x01FFFFFFull) | 0x80000000u }; // type=2
  const v8i g1 = {
    (int)((1u << 16)                 // data_size = 2 bytes
          | (1u << 20)               // pad_enable
          | (pad_iv << 22) | (pad_amt << 25)),
    (int)((td0 & 0xFFFFu) << 16),                         // [63:48] dim0.lo
    (int)(((td0 >> 16) & 0xFFFFu) | ((td1 & 0xFFFFu) << 16)),
    (int)(((td1 >> 16) & 0xFFFFu) | (tile_w << 16)),      // tile_dim0
    (int)(tile_h & 0xFFFFu),                              // tile_dim1
    (int)stride_elems,                                    // dim0_stride.lo
    0, 0 };
  const v4i gz = { 0, 0, 0, 0 };
#if __clang_major__ >= 23
  const v8i gz8 = { 0, 0, 0, 0, 0, 0, 0, 0 };
  __builtin_amdgcn_tensor_load_to_lds(g0, g1, gz, gz, gz8, 0);
#else
  __builtin_amdgcn_tensor_load_to_lds(g0, g1, gz, gz, 0);
#endif
}

// ---------------------------------------------------------------------------
// fp32 -> bf16 bulk convert (8 elems/thread; all sizes are multiples of 2048)
// ---------------------------------------------------------------------------
__global__ __launch_bounds__(256)
void cvt_bf16(const float* __restrict__ in, __bf16* __restrict__ out) {
  const size_t i = ((size_t)blockIdx.x * 256 + threadIdx.x) * 8;
  const float4 a = *(const float4*)(in + i);
  const float4 b = *(const float4*)(in + i + 4);
  v8bf o;
  o[0] = (__bf16)a.x; o[1] = (__bf16)a.y; o[2] = (__bf16)a.z; o[3] = (__bf16)a.w;
  o[4] = (__bf16)b.x; o[5] = (__bf16)b.y; o[6] = (__bf16)b.z; o[7] = (__bf16)b.w;
  *(v8bf*)(out + i) = o;
}

// ---------------------------------------------------------------------------
// GEMM: C[M,N](fp32) = A[M,K](bf16) * B[N,K](bf16)^T   (torch Linear)
// 256 thr (8 wave32), tile 128x128, BK=32, TDM double-buffered staging.
// Wave grid 4x2; each wave owns 32x64 = 2x4 WMMA tiles.
// ---------------------------------------------------------------------------
#define BM 128
#define BN 128
#define BK 32
#define ASTR 40   // LDS stride bf16: 64B row + 16B TDM pad  (16B-aligned frags)
#define BSTR 48   // LDS stride bf16: 64B row + 32B TDM pad  (32B-aligned v16bf)

__global__ __launch_bounds__(256)
void gemm_nt_wmma(const __bf16* __restrict__ A, const __bf16* __restrict__ B,
                  float* __restrict__ C, int M, int N, int K) {
  __shared__ __align__(128) __bf16 As[2][BM * ASTR];
  __shared__ __align__(128) __bf16 Bs[2][BN * BSTR];

  const int t    = threadIdx.x;
  const int lane = t & 31;
  const int wave = t >> 5;
  const int l16  = lane & 15;
  const int sub  = lane >> 4;
  const int wr   = wave & 3;             // 4 wave rows * 32
  const int wc   = wave >> 2;            // 2 wave cols * 64
  const int m0   = blockIdx.y * BM;
  const int n0   = blockIdx.x * BN;

  v8f acc[2][4];
  const v8f vzero = {0.f,0.f,0.f,0.f,0.f,0.f,0.f,0.f};
  for (int mt = 0; mt < 2; ++mt)
    for (int nt = 0; nt < 4; ++nt) acc[mt][nt] = vzero;

  const size_t abase = (size_t)m0 * K;
  const size_t bbase = (size_t)n0 * K;

  if (wave == 0) {   // prologue: stage tile 0
    tdm_load_2d(lds_off(&As[0][0]), A + abase, BK, BM, K, /*16dw*/3, /*4dw*/3);
    tdm_load_2d(lds_off(&Bs[0][0]), B + bbase, BK, BN, K, /*16dw*/3, /*8dw*/7);
  }

  const int nkt = K / BK;
  for (int kt = 0; kt < nkt; ++kt) {
    const int  cur  = kt & 1;
    const bool more = (kt + 1) < nkt;
    if (wave == 0 && more) {   // stage next tile into the other buffer
      tdm_load_2d(lds_off(&As[cur ^ 1][0]), A + abase + (size_t)(kt + 1) * BK,
                  BK, BM, K, 3, 3);
      tdm_load_2d(lds_off(&Bs[cur ^ 1][0]), B + bbase + (size_t)(kt + 1) * BK,
                  BK, BN, K, 3, 7);
    }
    if (more) __builtin_amdgcn_s_wait_tensorcnt((short)2);  // tile kt resident
    else      __builtin_amdgcn_s_wait_tensorcnt((short)0);
    __syncthreads();

    // A 16x32 frag: lanes0-15 row=l16 K={0..15}; lanes16-31 K={16..31}
    v16bf afrag[2];
    #pragma unroll
    for (int mt = 0; mt < 2; ++mt) {
      const int row = wr * 32 + mt * 16 + l16;
      const v8bf lo = *(const v8bf*)&As[cur][row * ASTR + sub * 16];
      const v8bf hi = *(const v8bf*)&As[cur][row * ASTR + sub * 16 + 8];
      afrag[mt] = __builtin_shufflevector(lo, hi, 0,1,2,3,4,5,6,7,
                                                  8,9,10,11,12,13,14,15);
    }
    // B 32x16 frag: lane = out col, contiguous K chunk sub*16
    v16bf bfrag[4];
    #pragma unroll
    for (int nt = 0; nt < 4; ++nt) {
      const int col = wc * 64 + nt * 16 + l16;
      bfrag[nt] = *(const v16bf*)&Bs[cur][col * BSTR + sub * 16];
    }
    #pragma unroll
    for (int mt = 0; mt < 2; ++mt)
      #pragma unroll
      for (int nt = 0; nt < 4; ++nt)
        acc[mt][nt] = __builtin_amdgcn_wmma_f32_16x16x32_bf16(
            false, afrag[mt], false, bfrag[nt], (short)0, acc[mt][nt],
            false, false);
    __syncthreads();   // all reads done -> buffer reusable by in-flight TDM
  }

  #pragma unroll
  for (int mt = 0; mt < 2; ++mt)
    #pragma unroll
    for (int nt = 0; nt < 4; ++nt) {
      const int col = n0 + wc * 64 + nt * 16 + l16;
      #pragma unroll
      for (int r = 0; r < 8; ++r) {
        const int row = m0 + wr * 32 + mt * 16 + sub * 8 + r;
        C[(size_t)row * N + col] = acc[mt][nt][r];
      }
    }
}

// ---------------------------------------------------------------------------
// RoPE on q: qraw [L, 2048] -> q_bf16 [NH, L, HD]
// ---------------------------------------------------------------------------
__global__ __launch_bounds__(256)
void rope_q(const float* __restrict__ qraw, __bf16* __restrict__ qb) {
  const int idx = blockIdx.x * 256 + threadIdx.x;   // NH*L*32
  const int i   = idx & 31;
  const int tmp = idx >> 5;
  const int l   = tmp & (LCTX - 1);
  const int h   = tmp >> 11;
  const float x0 = qraw[(size_t)l * DIM + h * HD + 2 * i];
  const float x1 = qraw[(size_t)l * DIM + h * HD + 2 * i + 1];
  const float theta = __powf(10000.0f, -2.0f * (float)i / 64.0f);
  float s, c;
  __sincosf((float)(l + 1) * theta, &s, &c);
  __bf16* dst = qb + ((size_t)h * LCTX + l) * HD + 2 * i;
  dst[0] = (__bf16)(c * x0 - s * x1);
  dst[1] = (__bf16)(s * x0 + c * x1);
}

// ---------------------------------------------------------------------------
// RoPE on k: kraw [L,512] -> k_bf16 [NKV,L,HD] + repeated fp32 d_out.k
// ---------------------------------------------------------------------------
__global__ __launch_bounds__(256)
void rope_k(const float* __restrict__ kraw, __bf16* __restrict__ kb,
            float* __restrict__ kout) {
  const int idx = blockIdx.x * 256 + threadIdx.x;   // NKV*L*32
  const int i   = idx & 31;
  const int tmp = idx >> 5;
  const int l   = tmp & (LCTX - 1);
  const int kvh = tmp >> 11;
  const float x0 = kraw[(size_t)l * (NKV * HD) + kvh * HD + 2 * i];
  const float x1 = kraw[(size_t)l * (NKV * HD) + kvh * HD + 2 * i + 1];
  const float theta = __powf(10000.0f, -2.0f * (float)i / 64.0f);
  float s, c;
  __sincosf((float)(l + 1) * theta, &s, &c);
  const float r0 = c * x0 - s * x1;
  const float r1 = s * x0 + c * x1;
  __bf16* dst = kb + ((size_t)kvh * LCTX + l) * HD + 2 * i;
  dst[0] = (__bf16)r0;
  dst[1] = (__bf16)r1;
  #pragma unroll
  for (int rr = 0; rr < 4; ++rr) {      // GQA repeat: h = kvh*4+rr
    float* o = kout + (((size_t)(kvh * 4 + rr) * LCTX) + l) * HD + 2 * i;
    o[0] = r0;
    o[1] = r1;
  }
}

// ---------------------------------------------------------------------------
// v: vraw [L,512] -> vt_bf16 [NKV,HD,L] (transposed) + repeated fp32 d_out.v
// ---------------------------------------------------------------------------
__global__ __launch_bounds__(256)
void prep_v(const float* __restrict__ vraw, __bf16* __restrict__ vtb,
            float* __restrict__ vout) {
  const int idx = blockIdx.x * 256 + threadIdx.x;   // NKV*L*HD
  const int d   = idx & 63;
  const int tmp = idx >> 6;
  const int l   = tmp & (LCTX - 1);
  const int kvh = tmp >> 11;
  const float val = vraw[(size_t)l * (NKV * HD) + kvh * HD + d];
  vtb[((size_t)kvh * HD + d) * LCTX + l] = (__bf16)val;
  #pragma unroll
  for (int rr = 0; rr < 4; ++rr)
    vout[(((size_t)(kvh * 4 + rr) * LCTX) + l) * HD + d] = val;
}

// ---------------------------------------------------------------------------
// Flash attention. Grid (L/128, NH); 8 waves, 16 query rows each; 64-key
// tiles TDM double-buffered in LDS. Output directly in bf16 [L, NH*HD].
// ---------------------------------------------------------------------------
#define KB 64
#define KSTR 80     // 128B row + 32B TDM pad -> 32B-aligned chunks
#define VSTR 80
#define PSTR 72     // per-wave P scratch, 16B-aligned chunks

__global__ __launch_bounds__(256)
void flash_attn(const __bf16* __restrict__ Qb, const __bf16* __restrict__ Kb,
                const __bf16* __restrict__ Vtb, __bf16* __restrict__ aoutb) {
  __shared__ __align__(128) __bf16 Ks[2][KB * KSTR];   // K tile  [key][64]
  __shared__ __align__(128) __bf16 Vts[2][HD * VSTR];  // Vt tile [dim][64 keys]
  __shared__ __align__(128) __bf16 Ps[8 * 16 * PSTR];  // per-wave P scratch

  const int h    = blockIdx.y;
  const int hk   = h >> 2;
  const int q0   = blockIdx.x * 128;
  const int t    = threadIdx.x;
  const int lane = t & 31, wave = t >> 5;
  const int l16  = lane & 15, sub = lane >> 4;

  // Q fragments (A layout) once from global
  const int qrow = q0 + wave * 16 + l16;
  const size_t qbase = ((size_t)h * LCTX + qrow) * HD;
  v16bf qfrag[2];
  #pragma unroll
  for (int ks = 0; ks < 2; ++ks) {
    const v8bf lo = *(const v8bf*)(Qb + qbase + ks * 32 + sub * 16);
    const v8bf hi = *(const v8bf*)(Qb + qbase + ks * 32 + sub * 16 + 8);
    qfrag[ks] = __builtin_shufflevector(lo, hi, 0,1,2,3,4,5,6,7,
                                                8,9,10,11,12,13,14,15);
  }

  const v8f vzero = {0.f,0.f,0.f,0.f,0.f,0.f,0.f,0.f};
  v8f oacc[4];
  #pragma unroll
  for (int nt = 0; nt < 4; ++nt) oacc[nt] = vzero;
  float mrow[8], lrow[8];
  #pragma unroll
  for (int r = 0; r < 8; ++r) { mrow[r] = -1e30f; lrow[r] = 0.0f; }

  __bf16* myP = &Ps[wave * 16 * PSTR];
  const int nkb = (q0 + 128) / KB;     // causal block count

  if (wave == 0) {   // prologue: stage key-block 0
    tdm_load_2d(lds_off(&Ks[0][0]),  Kb  + ((size_t)hk * LCTX) * HD,
                HD, KB, HD, /*32dw*/4, /*8dw*/7);
    tdm_load_2d(lds_off(&Vts[0][0]), Vtb + ((size_t)hk * HD) * LCTX,
                KB, HD, LCTX, 4, 7);
  }

  for (int kb = 0; kb < nkb; ++kb) {
    const int  k0   = kb * KB;
    const int  cur  = kb & 1;
    const bool more = (kb + 1) < nkb;
    if (wave == 0 && more) {
      tdm_load_2d(lds_off(&Ks[cur ^ 1][0]),
                  Kb + ((size_t)hk * LCTX + k0 + KB) * HD, HD, KB, HD, 4, 7);
      tdm_load_2d(lds_off(&Vts[cur ^ 1][0]),
                  Vtb + ((size_t)hk * HD) * LCTX + k0 + KB, KB, HD, LCTX, 4, 7);
    }
    if (more) __builtin_amdgcn_s_wait_tensorcnt((short)2);
    else      __builtin_amdgcn_s_wait_tensorcnt((short)0);
    __syncthreads();

    // ---- S = Q K^T : 4 key-column tiles of 16 ----------------------------
    v8f s[4];
    #pragma unroll
    for (int nt = 0; nt < 4; ++nt) {
      s[nt] = vzero;
      #pragma unroll
      for (int ks = 0; ks < 2; ++ks) {
        const v16bf kf = *(const v16bf*)&Ks[cur][(nt * 16 + l16) * KSTR +
                                                 ks * 32 + sub * 16];
        s[nt] = __builtin_amdgcn_wmma_f32_16x16x32_bf16(
            false, qfrag[ks], false, kf, (short)0, s[nt], false, false);
      }
    }

    // ---- scale + causal mask (C layout: row=sub*8+r, col=l16) ------------
    const int rbase = q0 + wave * 16 + sub * 8;
    #pragma unroll
    for (int nt = 0; nt < 4; ++nt) {
      const int key = k0 + nt * 16 + l16;
      #pragma unroll
      for (int r = 0; r < 8; ++r) {
        const float v = s[nt][r] * SCALE;
        s[nt][r] = (key > rbase + r) ? -1e30f : v;
      }
    }

    // ---- online softmax ---------------------------------------------------
    float tmax[8];
    #pragma unroll
    for (int r = 0; r < 8; ++r) {
      tmax[r] = fmaxf(fmaxf(s[0][r], s[1][r]), fmaxf(s[2][r], s[3][r]));
      #pragma unroll
      for (int off = 8; off >= 1; off >>= 1)
        tmax[r] = fmaxf(tmax[r], __shfl_xor(tmax[r], off, 32));
    }
    float alpha[8], tsum[8];
    #pragma unroll
    for (int r = 0; r < 8; ++r) {
      const float mnew = fmaxf(mrow[r], tmax[r]);
      alpha[r] = __expf(mrow[r] - mnew);
      mrow[r] = mnew;
      tsum[r] = 0.0f;
    }
    #pragma unroll
    for (int nt = 0; nt < 4; ++nt)
      #pragma unroll
      for (int r = 0; r < 8; ++r) {
        const float p = __expf(s[nt][r] - mrow[r]);
        tsum[r] += p;
        myP[(sub * 8 + r) * PSTR + nt * 16 + l16] = (__bf16)p;  // LDS transpose
      }
    #pragma unroll
    for (int r = 0; r < 8; ++r) {
      #pragma unroll
      for (int off = 8; off >= 1; off >>= 1)
        tsum[r] += __shfl_xor(tsum[r], off, 32);
      lrow[r] = alpha[r] * lrow[r] + tsum[r];
    }
    #pragma unroll
    for (int nt = 0; nt < 4; ++nt)
      #pragma unroll
      for (int r = 0; r < 8; ++r) oacc[nt][r] *= alpha[r];

    asm volatile("s_wait_dscnt 0" ::: "memory");   // same-wave P write->read

    // ---- O += P V ---------------------------------------------------------
    v16bf pfrag[2];
    #pragma unroll
    for (int ks = 0; ks < 2; ++ks) {
      const v8bf lo = *(const v8bf*)&myP[l16 * PSTR + ks * 32 + sub * 16];
      const v8bf hi = *(const v8bf*)&myP[l16 * PSTR + ks * 32 + sub * 16 + 8];
      pfrag[ks] = __builtin_shufflevector(lo, hi, 0,1,2,3,4,5,6,7,
                                                  8,9,10,11,12,13,14,15);
    }
    #pragma unroll
    for (int nt = 0; nt < 4; ++nt)
      #pragma unroll
      for (int ks = 0; ks < 2; ++ks) {
        const v16bf vf = *(const v16bf*)&Vts[cur][(nt * 16 + l16) * VSTR +
                                                  ks * 32 + sub * 16];
        oacc[nt] = __builtin_amdgcn_wmma_f32_16x16x32_bf16(
            false, pfrag[ks], false, vf, (short)0, oacc[nt], false, false);
      }
    __syncthreads();   // buffer reusable by in-flight TDM
  }

  // ---- finalize: O /= l, write bf16 attn_out [L, NH*HD] -------------------
  #pragma unroll
  for (int nt = 0; nt < 4; ++nt) {
    const int col = h * HD + nt * 16 + l16;
    #pragma unroll
    for (int r = 0; r < 8; ++r) {
      const int row = q0 + wave * 16 + sub * 8 + r;
      aoutb[(size_t)row * DIM + col] = (__bf16)(oacc[nt][r] / lrow[r]);
    }
  }
}

// ---------------------------------------------------------------------------
// Host launcher
// ---------------------------------------------------------------------------
extern "C" void kernel_launch(void* const* d_in, const int* in_sizes, int n_in,
                              void* d_out, int out_size, void* d_ws, size_t ws_size,
                              hipStream_t stream) {
  const float* x  = (const float*)d_in[0];
  // d_in[1] = mask (unused; causal handled analytically)
  const float* wq = (const float*)d_in[2];
  const float* wk = (const float*)d_in[3];
  const float* wv = (const float*)d_in[4];
  const float* wo = (const float*)d_in[5];

  float* outp = (float*)d_out;                       // [L, DIM]
  float* kout = outp + (size_t)LCTX * DIM;           // [NH, L, HD]
  float* vout = kout + (size_t)NH * LCTX * HD;       // [NH, L, HD]

  char* w = (char*)d_ws;
  float*  qraw = (float*)w;   w += (size_t)LCTX * DIM * 4;
  float*  kraw = (float*)w;   w += (size_t)LCTX * NKV * HD * 4;
  float*  vraw = (float*)w;   w += (size_t)LCTX * NKV * HD * 4;
  __bf16* xb   = (__bf16*)w;  w += (size_t)LCTX * DIM * 2;
  __bf16* wqb  = (__bf16*)w;  w += (size_t)DIM * DIM * 2;
  __bf16* wkb  = (__bf16*)w;  w += (size_t)NKV * HD * DIM * 2;
  __bf16* wvb  = (__bf16*)w;  w += (size_t)NKV * HD * DIM * 2;
  __bf16* wob  = (__bf16*)w;  w += (size_t)DIM * DIM * 2;
  __bf16* qb   = (__bf16*)w;  w += (size_t)NH * LCTX * HD * 2;
  __bf16* kb   = (__bf16*)w;  w += (size_t)NKV * LCTX * HD * 2;
  __bf16* vtb  = (__bf16*)w;  w += (size_t)NKV * HD * LCTX * 2;
  __bf16* aob  = (__bf16*)w;  w += (size_t)LCTX * DIM * 2;

  dim3 blk(256);

  // one-shot fp32 -> bf16 conversions (bandwidth-trivial: ~56MB @ 23.3 TB/s)
  cvt_bf16<<<(LCTX * DIM) / 2048, blk, 0, stream>>>(x,  xb);
  cvt_bf16<<<(DIM * DIM) / 2048, blk, 0, stream>>>(wq, wqb);
  cvt_bf16<<<(NKV * HD * DIM) / 2048, blk, 0, stream>>>(wk, wkb);
  cvt_bf16<<<(NKV * HD * DIM) / 2048, blk, 0, stream>>>(wv, wvb);
  cvt_bf16<<<(DIM * DIM) / 2048, blk, 0, stream>>>(wo, wob);

  // projections: y = x @ W^T (fp32 out for RoPE precision)
  gemm_nt_wmma<<<dim3(DIM / BN, LCTX / BM), blk, 0, stream>>>(xb, wqb, qraw, LCTX, DIM, DIM);
  gemm_nt_wmma<<<dim3((NKV * HD) / BN, LCTX / BM), blk, 0, stream>>>(xb, wkb, kraw, LCTX, NKV * HD, DIM);
  gemm_nt_wmma<<<dim3((NKV * HD) / BN, LCTX / BM), blk, 0, stream>>>(xb, wvb, vraw, LCTX, NKV * HD, DIM);

  // RoPE + GQA repeat + operand layout prep
  rope_q<<<(NH * LCTX * 32) / 256, blk, 0, stream>>>(qraw, qb);
  rope_k<<<(NKV * LCTX * 32) / 256, blk, 0, stream>>>(kraw, kb, kout);
  prep_v<<<(NKV * LCTX * HD) / 256, blk, 0, stream>>>(vraw, vtb, vout);

  // flash attention -> bf16 activations for the out-projection
  flash_attn<<<dim3(LCTX / 128, NH), blk, 0, stream>>>(qb, kb, vtb, aob);

  // out = attn @ wo^T
  gemm_nt_wmma<<<dim3(DIM / BN, LCTX / BM), blk, 0, stream>>>(aob, wob, outp, LCTX, DIM, DIM);
}
